// YOLOLayoutDetector_77988016161156
// MI455X (gfx1250) — matrix operand order repaired and checked by reading, not compiled, not run on hardware
//
#include <hip/hip_runtime.h>
#include <cstdint>
#include <cstddef>

// ---------------------------------------------------------------------------
// Types for CDNA5 WMMA (gfx1250, wave32)
// ---------------------------------------------------------------------------
typedef __attribute__((ext_vector_type(16))) __bf16 v16bf;
typedef __attribute__((ext_vector_type(8)))  float  v8f;

#define BATCH 8
#define NTOT  5376          // 64*64 + 32*32 + 16*16 anchors per image
#define CONF_T 0.25f
#define IOU_T  0.45f

// float -> bf16 (round-to-nearest-even)
static __device__ __forceinline__ unsigned short f2bf(float f) {
    unsigned u = __builtin_bit_cast(unsigned, f);
    unsigned r = ((u >> 16) & 1u) + 0x7FFFu;
    return (unsigned short)((u + r) >> 16);
}

// WMMA fragment: 8 dwords = two 16B groups (A: words base+4h..+3 / base+8+4h..+3;
// B: same words as A). Loaded as uint4 pairs.
union Frag { v16bf v; uint4 q[2]; unsigned u[8]; };

// ---------------------------------------------------------------------------
// TDM / cluster probe (compiled into the bundle, never launched): exercises
// tensor_load_to_lds + s_wait_tensorcnt + cluster barrier codegen paths.
// ---------------------------------------------------------------------------
#if defined(__has_builtin)
#if __has_builtin(__builtin_amdgcn_tensor_load_to_lds)
typedef unsigned tdm_v4u __attribute__((ext_vector_type(4)));
typedef int      tdm_v8i __attribute__((ext_vector_type(8)));
typedef int      tdm_v4i __attribute__((ext_vector_type(4)));
__global__ void tdm_probe_kernel() {
    tdm_v4u g0 = {};
    tdm_v8i g1 = {};
    tdm_v4i g2 = {};
    tdm_v4i g3 = {};
#if __clang_major__ >= 23
    tdm_v8i g4 = {};
    __builtin_amdgcn_tensor_load_to_lds(g0, g1, g2, g3, g4, 0);
#else
    __builtin_amdgcn_tensor_load_to_lds(g0, g1, g2, g3, 0);
#endif
    __builtin_amdgcn_s_wait_tensorcnt(0);
#if __has_builtin(__builtin_amdgcn_s_cluster_barrier)
    __builtin_amdgcn_s_cluster_barrier();
#endif
}
#endif
#endif

// ---------------------------------------------------------------------------
// Weight prep: fold BN scale g into W, convert to bf16, layout [Cout][K]
// with K-order = (ky*3+kx)*Cin + ic  (so a 32-wide K chunk stays in one tap)
// ---------------------------------------------------------------------------
__global__ void prep_w3x3(const float* __restrict__ w, const float* __restrict__ g,
                          unsigned short* __restrict__ wbf, int Cin) {
    const int K = Cin * 9;
    const int total = 256 * K;
    for (int t = blockIdx.x * blockDim.x + threadIdx.x; t < total;
         t += gridDim.x * blockDim.x) {
        int oc = t / K, kk = t - oc * K;
        int kpos = kk / Cin, ic = kk - kpos * Cin;
        float v = w[(size_t)(oc * Cin + ic) * 9 + kpos] * g[oc];
        wbf[(size_t)oc * K + kk] = f2bf(v);
    }
}

// 1x1 head: [77,256] -> bf16 [80][256], rows 77..79 zero-padded
__global__ void prep_whead(const float* __restrict__ w, unsigned short* __restrict__ wbf) {
    int t = blockIdx.x * blockDim.x + threadIdx.x;
    if (t >= 80 * 256) return;
    int oc = t >> 8, ic = t & 255;
    float v = (oc < 77) ? w[oc * 256 + ic] : 0.f;
    wbf[t] = f2bf(v);
}

// ---------------------------------------------------------------------------
// 3x3 conv as WMMA GEMM. One 256-thread block (8 waves) owns a 16-pixel tile:
// the 16x32 bf16 im2col chunk is cooperatively staged into LDS once per
// K-step, then each wave consumes it (ds_load_b128) against 2 out-channel
// tiles (8 waves x 2 = 256 channels). A fragments are 2x b128 global loads.
// ---------------------------------------------------------------------------
#define BROW 20   // LDS row stride in dwords (16 K-pairs + 4 pad, bank spread)

__global__ __launch_bounds__(256)
void conv3x3_f32in(const float* __restrict__ in, const unsigned short* __restrict__ wbf,
                   const float* __restrict__ bias, unsigned short* __restrict__ out,
                   int Cin, int H, int W) {
    __shared__ unsigned bstage[16 * BROW];   // [pixel][K/2] packed bf16x2

    const int tid  = threadIdx.x;
    const int wave = tid >> 5;
    const int lane = tid & 31;
    const int half = lane >> 4;
    const int l16  = lane & 15;
    const int colBase = blockIdx.x << 4;
    const int HW = H * W;
    const int K  = Cin * 9;

    // this lane's output pixel
    const int n    = colBase + l16;
    const int bimg = n / HW;
    const int rem  = n - bimg * HW;

    // staging assignment: thread -> (pixel sn, K-pair sk)
    const int sn = tid >> 4;            // 0..15
    const int sk = (tid & 15) << 1;     // 0,2,..,30
    const int gn = colBase + sn;
    const int gb = gn / HW;
    const int grem = gn - gb * HW;
    const int gy = grem / W;
    const int gx = grem - gy * W;

    const int m0 = (wave << 5) + l16;        // A rows, tile 0
    const int m1 = m0 + 16;                  // A rows, tile 1

    v8f acc0 = {}, acc1 = {};
    for (int kpos = 0; kpos < 9; ++kpos) {
        const int ky = kpos / 3, kx = kpos - 3 * (kpos / 3);
        const int sy = gy + ky - 1, sx = gx + kx - 1;
        const bool sinb = (sy >= 0) & (sy < H) & (sx >= 0) & (sx < W);
        const float* sbase = in + (size_t)gb * Cin * HW + (size_t)sy * W + sx;

        for (int icb = 0; icb < Cin; icb += 32) {
            // --- cooperative stage of B chunk (16 px x 32 K) into LDS ---
            float f0 = sinb ? sbase[(size_t)(icb + sk) * HW] : 0.f;
            float f1 = sinb ? sbase[(size_t)(icb + sk + 1) * HW] : 0.f;
            bstage[sn * BROW + (sk >> 1)] =
                (unsigned)f2bf(f0) | ((unsigned)f2bf(f1) << 16);
            __syncthreads();

            // --- A fragments: two b128 loads per tile ---
            const int kkrow = kpos * Cin + icb;
            __builtin_prefetch(wbf + (size_t)m0 * K + kkrow + 32, 0, 1);
            Frag A0, A1, Bm;
            const uint4* wq0 = (const uint4*)(wbf + (size_t)m0 * K + kkrow);
            const uint4* wq1 = (const uint4*)(wbf + (size_t)m1 * K + kkrow);
            A0.q[0] = wq0[half];     A0.q[1] = wq0[2 + half];
            A1.q[0] = wq1[half];     A1.q[1] = wq1[2 + half];

            // --- B fragment: two b128 LDS loads ---
            const uint4* lq = (const uint4*)(bstage + (size_t)l16 * BROW);
            Bm.q[0] = lq[half];
            Bm.q[1] = lq[2 + half];

            acc0 = __builtin_amdgcn_wmma_f32_16x16x32_bf16(
                false, A0.v, false, Bm.v, (short)0, acc0, false, false);
            acc1 = __builtin_amdgcn_wmma_f32_16x16x32_bf16(
                false, A1.v, false, Bm.v, (short)0, acc1, false, false);
            __syncthreads();
        }
    }
#pragma unroll
    for (int r = 0; r < 8; ++r) {
        int moA = (wave << 5) + r + half * 8;
        int moB = moA + 16;
        float vA = acc0[r] + bias[moA];
        float vB = acc1[r] + bias[moB];
        vA = vA > 0.f ? vA : 0.f;
        vB = vB > 0.f ? vB : 0.f;
        out[(size_t)(bimg * 256 + moA) * HW + rem] = f2bf(vA);
        out[(size_t)(bimg * 256 + moB) * HW + rem] = f2bf(vB);
    }
}

// Same structure, bf16 input (second tower layer), Cin = 256.
__global__ __launch_bounds__(256)
void conv3x3_bf16in(const unsigned short* __restrict__ in,
                    const unsigned short* __restrict__ wbf,
                    const float* __restrict__ bias, unsigned short* __restrict__ out,
                    int H, int W) {
    __shared__ unsigned bstage[16 * BROW];

    const int tid  = threadIdx.x;
    const int wave = tid >> 5;
    const int lane = tid & 31;
    const int half = lane >> 4;
    const int l16  = lane & 15;
    const int colBase = blockIdx.x << 4;
    const int HW = H * W;
    const int Cin = 256, K = 256 * 9;

    const int n    = colBase + l16;
    const int bimg = n / HW;
    const int rem  = n - bimg * HW;

    const int sn = tid >> 4;
    const int sk = (tid & 15) << 1;
    const int gn = colBase + sn;
    const int gb = gn / HW;
    const int grem = gn - gb * HW;
    const int gy = grem / W;
    const int gx = grem - gy * W;

    const int m0 = (wave << 5) + l16;
    const int m1 = m0 + 16;

    v8f acc0 = {}, acc1 = {};
    for (int kpos = 0; kpos < 9; ++kpos) {
        const int ky = kpos / 3, kx = kpos - 3 * (kpos / 3);
        const int sy = gy + ky - 1, sx = gx + kx - 1;
        const bool sinb = (sy >= 0) & (sy < H) & (sx >= 0) & (sx < W);
        const unsigned short* sbase =
            in + (size_t)gb * Cin * HW + (size_t)sy * W + sx;

        for (int icb = 0; icb < Cin; icb += 32) {
            unsigned short u0 = sinb ? sbase[(size_t)(icb + sk) * HW] : (unsigned short)0;
            unsigned short u1 = sinb ? sbase[(size_t)(icb + sk + 1) * HW] : (unsigned short)0;
            bstage[sn * BROW + (sk >> 1)] = (unsigned)u0 | ((unsigned)u1 << 16);
            __syncthreads();

            const int kkrow = kpos * Cin + icb;
            __builtin_prefetch(wbf + (size_t)m0 * K + kkrow + 32, 0, 1);
            Frag A0, A1, Bm;
            const uint4* wq0 = (const uint4*)(wbf + (size_t)m0 * K + kkrow);
            const uint4* wq1 = (const uint4*)(wbf + (size_t)m1 * K + kkrow);
            A0.q[0] = wq0[half];     A0.q[1] = wq0[2 + half];
            A1.q[0] = wq1[half];     A1.q[1] = wq1[2 + half];

            const uint4* lq = (const uint4*)(bstage + (size_t)l16 * BROW);
            Bm.q[0] = lq[half];
            Bm.q[1] = lq[2 + half];

            acc0 = __builtin_amdgcn_wmma_f32_16x16x32_bf16(
                false, A0.v, false, Bm.v, (short)0, acc0, false, false);
            acc1 = __builtin_amdgcn_wmma_f32_16x16x32_bf16(
                false, A1.v, false, Bm.v, (short)0, acc1, false, false);
            __syncthreads();
        }
    }
#pragma unroll
    for (int r = 0; r < 8; ++r) {
        int moA = (wave << 5) + r + half * 8;
        int moB = moA + 16;
        float vA = acc0[r] + bias[moA];
        float vB = acc1[r] + bias[moB];
        vA = vA > 0.f ? vA : 0.f;
        vB = vB > 0.f ? vB : 0.f;
        out[(size_t)(bimg * 256 + moA) * HW + rem] = f2bf(vA);
        out[(size_t)(bimg * 256 + moB) * HW + rem] = f2bf(vB);
    }
}

// 1x1 head conv: bf16 act [B,256,H,W] x bf16 W[80,256] -> f32 pred [B,80,H,W]
// Small (~1.6 GFLOP total): one wave per 16x16 tile, b128 A loads.
__global__ __launch_bounds__(32)
void head1x1(const unsigned short* __restrict__ in, const unsigned short* __restrict__ wbf,
             const float* __restrict__ b3, float* __restrict__ pred, int H, int W) {
    const int lane = threadIdx.x;
    const int half = lane >> 4;
    const int l16  = lane & 15;
    const int rowBase = blockIdx.y << 4;   // 0..79 in tiles of 16
    const int colBase = blockIdx.x << 4;
    const int HW = H * W;

    const int n    = colBase + l16;
    const int bimg = n / HW;
    const int rem  = n - bimg * HW;
    const int m = rowBase + l16;

    v8f acc = {};
    for (int icb = 0; icb < 256; icb += 32) {
        Frag A, Bm;
        const uint4* wq = (const uint4*)(wbf + (size_t)m * 256 + icb);
        A.q[0] = wq[half];
        A.q[1] = wq[2 + half];
#pragma unroll
        for (int v = 0; v < 8; ++v) {
            int ic0 = icb + half * 8 + 2 * v + (v >= 4 ? 8 : 0);
            unsigned short u0 = in[(size_t)(bimg * 256 + ic0) * HW + rem];
            unsigned short u1 = in[(size_t)(bimg * 256 + ic0 + 1) * HW + rem];
            Bm.u[v] = (unsigned)u0 | ((unsigned)u1 << 16);
        }
        acc = __builtin_amdgcn_wmma_f32_16x16x32_bf16(
            false, A.v, false, Bm.v, (short)0, acc, false, false);
    }
#pragma unroll
    for (int r = 0; r < 8; ++r) {
        int mo = rowBase + r + half * 8;
        float bv = (mo < 77) ? b3[mo] : 0.f;
        pred[(size_t)(bimg * 80 + mo) * HW + rem] = acc[r] + bv;
    }
}

// ---------------------------------------------------------------------------
// DFL decode + class sigmoid max/argmax per pixel
// ---------------------------------------------------------------------------
__global__ void decode_k(const float* __restrict__ pred, float* __restrict__ boxes,
                         float* __restrict__ scores, int* __restrict__ classes,
                         int H, int W, int baseIdx) {
    const int HW = H * W;
    int t = blockIdx.x * blockDim.x + threadIdx.x;
    if (t >= BATCH * HW) return;
    int b = t / HW, rem = t - b * HW;
    int y = rem / W, x = rem - y * W;
    const float* pp = pred + (size_t)b * 80 * HW + rem;   // channel stride = HW

    float dist[4];
#pragma unroll
    for (int s4 = 0; s4 < 4; ++s4) {
        float l[16], mx = -1e30f;
#pragma unroll
        for (int j = 0; j < 16; ++j) { l[j] = pp[(size_t)(s4 * 16 + j) * HW]; mx = fmaxf(mx, l[j]); }
        float se = 0.f, sw = 0.f;
#pragma unroll
        for (int j = 0; j < 16; ++j) { float e = __expf(l[j] - mx); se += e; sw += e * (float)j; }
        dist[s4] = sw / se;
    }
    float x1 = (float)x - dist[0], y1 = (float)y - dist[1];
    float x2 = (float)x + dist[2], y2 = (float)y + dist[3];

    float best = -1e30f; int bi = 0;
#pragma unroll
    for (int c = 0; c < 13; ++c) {
        float v = pp[(size_t)(64 + c) * HW];
        if (v > best) { best = v; bi = c; }
    }
    float sc = 1.f / (1.f + __expf(-best));

    int g = b * NTOT + baseIdx + rem;
    boxes[(size_t)g * 4 + 0] = x1;
    boxes[(size_t)g * 4 + 1] = y1;
    boxes[(size_t)g * 4 + 2] = x2;
    boxes[(size_t)g * 4 + 3] = y2;
    scores[g]  = sc;
    classes[g] = bi;
}

// ---------------------------------------------------------------------------
// Descending stable sort (matching argsort(where(valid,s,-inf))[::-1]) by
// O(n^2) rank computation; one block per image.
// ---------------------------------------------------------------------------
__global__ __launch_bounds__(256)
void sortrank(const float* __restrict__ scores, int* __restrict__ sidx) {
    __shared__ float key[NTOT];
    const int b = blockIdx.x;
    const float* s = scores + (size_t)b * NTOT;
    for (int i = threadIdx.x; i < NTOT; i += blockDim.x)
        key[i] = (s[i] > CONF_T) ? s[i] : -__builtin_inff();
    __syncthreads();
    for (int i = threadIdx.x; i < NTOT; i += blockDim.x) {
        float ki = key[i];
        int r = 0;
        for (int j = 0; j < NTOT; ++j) {
            float kj = key[j];
            r += (kj > ki) | ((kj == ki) & (j > i));  // ties: larger idx first
        }
        sidx[b * NTOT + r] = i;
    }
}

__global__ void gathersorted(const int* __restrict__ sidx, const float* __restrict__ boxes,
                             const float* __restrict__ scores, const int* __restrict__ classes,
                             float* __restrict__ sb, float* __restrict__ ss,
                             int* __restrict__ scl, unsigned char* __restrict__ sv) {
    int t = blockIdx.x * blockDim.x + threadIdx.x;
    if (t >= BATCH * NTOT) return;
    int b = t / NTOT;
    int si = sidx[t];
    int g = b * NTOT + si;
#pragma unroll
    for (int k = 0; k < 4; ++k) sb[(size_t)t * 4 + k] = boxes[(size_t)g * 4 + k];
    ss[t]  = scores[g];
    scl[t] = classes[g];
    sv[t]  = scores[g] > CONF_T ? 1 : 0;
}

// Greedy per-class NMS; one block per image, sequential over i, parallel j.
__global__ __launch_bounds__(256)
void nms_k(const float* __restrict__ sb, const int* __restrict__ scl,
           const unsigned char* __restrict__ sv, unsigned char* __restrict__ keep) {
    __shared__ unsigned char sup[NTOT];
    __shared__ float cb[4];
    __shared__ int ccl, ccur;
    const int b = blockIdx.x;
    const float* boxes = sb + (size_t)b * NTOT * 4;
    const int* cls = scl + b * NTOT;
    for (int i = threadIdx.x; i < NTOT; i += blockDim.x) sup[i] = 0;
    __syncthreads();
    for (int i = 0; i < NTOT; ++i) {
        if (threadIdx.x == 0) {
            int cur = sv[b * NTOT + i] && !sup[i];
            keep[b * NTOT + i] = (unsigned char)cur;
            ccur = cur;
            cb[0] = boxes[(size_t)i * 4 + 0]; cb[1] = boxes[(size_t)i * 4 + 1];
            cb[2] = boxes[(size_t)i * 4 + 2]; cb[3] = boxes[(size_t)i * 4 + 3];
            ccl = cls[i];
        }
        __syncthreads();
        if (ccur) {
            float a1 = (cb[2] - cb[0]) * (cb[3] - cb[1]);
            for (int j = i + 1 + threadIdx.x; j < NTOT; j += blockDim.x) {
                if (cls[j] != ccl) continue;
                float bx1 = boxes[(size_t)j * 4 + 0], by1 = boxes[(size_t)j * 4 + 1];
                float bx2 = boxes[(size_t)j * 4 + 2], by2 = boxes[(size_t)j * 4 + 3];
                float xx1 = fmaxf(cb[0], bx1), yy1 = fmaxf(cb[1], by1);
                float xx2 = fminf(cb[2], bx2), yy2 = fminf(cb[3], by2);
                float iw = xx2 - xx1, ih = yy2 - yy1;
                float inter = (iw < 0.f || ih < 0.f) ? 0.f : iw * ih;
                float a2 = (bx2 - bx1) * (by2 - by1);
                float un = a1 + a2 - inter;
                float iou = un > 0.f ? inter / un : 0.f;
                if (iou > IOU_T) sup[j] = 1;
            }
        }
        __syncthreads();
    }
}

// out = concat([boxes, score],-1)*keep  |  cl (float)  |  keep (float)
__global__ void writeout(const float* __restrict__ sb, const float* __restrict__ ss,
                         const int* __restrict__ scl, const unsigned char* __restrict__ keep,
                         float* __restrict__ out) {
    int t = blockIdx.x * blockDim.x + threadIdx.x;
    if (t >= BATCH * NTOT) return;
    float kf = keep[t] ? 1.f : 0.f;
#pragma unroll
    for (int k = 0; k < 4; ++k) out[(size_t)t * 5 + k] = sb[(size_t)t * 4 + k] * kf;
    out[(size_t)t * 5 + 4] = ss[t] * kf;
    out[(size_t)BATCH * NTOT * 5 + t] = (float)scl[t];
    out[(size_t)BATCH * NTOT * 6 + t] = kf;
}

// ---------------------------------------------------------------------------
extern "C" void kernel_launch(void* const* d_in, const int* in_sizes, int n_in,
                              void* d_out, int out_size, void* d_ws, size_t ws_size,
                              hipStream_t stream) {
    (void)in_sizes; (void)n_in; (void)out_size; (void)ws_size;
    const float* p[3] = {(const float*)d_in[0], (const float*)d_in[1], (const float*)d_in[2]};
    struct Sc { const float *w1,*g1,*b1,*w2,*g2,*b2,*w3,*b3; } sc[3];
    for (int s = 0; s < 3; ++s) {
        int b = 3 + s * 8;
        sc[s] = { (const float*)d_in[b + 0], (const float*)d_in[b + 1],
                  (const float*)d_in[b + 2], (const float*)d_in[b + 3],
                  (const float*)d_in[b + 4], (const float*)d_in[b + 5],
                  (const float*)d_in[b + 6], (const float*)d_in[b + 7] };
    }
    static const int CIN[3] = {192, 384, 768};
    static const int HH[3]  = {64, 32, 16};
    static const int WW[3]  = {64, 32, 16};
    static const int BASE[3]= {0, 4096, 5120};

    char* ws = (char*)d_ws;
    size_t off = 0;
    auto alloc = [&](size_t bytes) -> char* {
        off = (off + 255) & ~(size_t)255;
        char* ptr = ws + off;
        off += bytes;
        return ptr;
    };
    unsigned short *w1bf[3], *w2bf[3], *w3bf[3];
    for (int s = 0; s < 3; ++s) {
        w1bf[s] = (unsigned short*)alloc((size_t)256 * CIN[s] * 9 * 2);
        w2bf[s] = (unsigned short*)alloc((size_t)256 * 256 * 9 * 2);
        w3bf[s] = (unsigned short*)alloc((size_t)80 * 256 * 2);
    }
    unsigned short* act1 = (unsigned short*)alloc((size_t)BATCH * 256 * 64 * 64 * 2);
    unsigned short* act2 = (unsigned short*)alloc((size_t)BATCH * 256 * 64 * 64 * 2);
    float* pred   = (float*)alloc((size_t)BATCH * 80 * 64 * 64 * 4);
    float* boxes  = (float*)alloc((size_t)BATCH * NTOT * 4 * 4);
    float* scores = (float*)alloc((size_t)BATCH * NTOT * 4);
    int*   classes= (int*)  alloc((size_t)BATCH * NTOT * 4);
    int*   sidx   = (int*)  alloc((size_t)BATCH * NTOT * 4);
    float* sb     = (float*)alloc((size_t)BATCH * NTOT * 4 * 4);
    float* ssc    = (float*)alloc((size_t)BATCH * NTOT * 4);
    int*   scl    = (int*)  alloc((size_t)BATCH * NTOT * 4);
    unsigned char* sv   = (unsigned char*)alloc((size_t)BATCH * NTOT);
    unsigned char* keep = (unsigned char*)alloc((size_t)BATCH * NTOT);

    // Weight prep (BN fold + bf16 pack, GEMM-friendly K order)
    for (int s = 0; s < 3; ++s) {
        int n1 = 256 * CIN[s] * 9;
        prep_w3x3<<<(n1 + 255) / 256, 256, 0, stream>>>(sc[s].w1, sc[s].g1, w1bf[s], CIN[s]);
        int n2 = 256 * 256 * 9;
        prep_w3x3<<<(n2 + 255) / 256, 256, 0, stream>>>(sc[s].w2, sc[s].g2, w2bf[s], 256);
        prep_whead<<<(80 * 256 + 255) / 256, 256, 0, stream>>>(sc[s].w3, w3bf[s]);
    }

    // Towers + decode, one scale at a time (act/pred buffers reused)
    for (int s = 0; s < 3; ++s) {
        int H = HH[s], W = WW[s], HW = H * W, tot = BATCH * HW;
        dim3 gConv(tot / 16);   // 8 waves per block cover all 256 out-channels
        conv3x3_f32in <<<gConv, 256, 0, stream>>>(p[s], w1bf[s], sc[s].b1, act1, CIN[s], H, W);
        conv3x3_bf16in<<<gConv, 256, 0, stream>>>(act1, w2bf[s], sc[s].b2, act2, H, W);
        dim3 gHead(tot / 16, 5);
        head1x1<<<gHead, 32, 0, stream>>>(act2, w3bf[s], sc[s].b3, pred, H, W);
        decode_k<<<(tot + 255) / 256, 256, 0, stream>>>(pred, boxes, scores, classes, H, W, BASE[s]);
    }

    // Sort + NMS + output
    sortrank<<<BATCH, 256, 0, stream>>>(scores, sidx);
    gathersorted<<<(BATCH * NTOT + 255) / 256, 256, 0, stream>>>(sidx, boxes, scores, classes,
                                                                 sb, ssc, scl, sv);
    nms_k<<<BATCH, 256, 0, stream>>>(sb, scl, sv, keep);
    writeout<<<(BATCH * NTOT + 255) / 256, 256, 0, stream>>>(sb, ssc, scl, keep, (float*)d_out);
}
